// CLIT_26980984553502
// MI455X (gfx1250) — compile-verified, matrix operand order checked by reading, not run on hardware
//
#include <hip/hip_runtime.h>
#include <hip/hip_bf16.h>
#include <stdint.h>

// ---------------------------------------------------------------------------
// CLIT local-attention SR pipeline for gfx1250 (MI455X), wave32 + WMMA.
// All GEMM-like work (ch/q/k/v convs, MLP m0..m3) runs on
// v_wmma_f32_16x16x32_f16. Weights are pre-repacked into per-lane fragment
// order so B operands are contiguous 32B loads; A operands are staged in LDS.
// ---------------------------------------------------------------------------

typedef __attribute__((ext_vector_type(16))) _Float16     v16h;
typedef __attribute__((ext_vector_type(8)))  float        v8f;
typedef __attribute__((ext_vector_type(4)))  unsigned int u32x4;

#define B_    2
#define Hh    64
#define Ww    64
#define P_    (Hh*Ww)
#define Qn    4096
#define DIMc  192
#define HEADS 8
#define HD    24
#define RAD   3
#define RRr   7
#define RAx   49
#define ENC   64
#define HID   256

union FragH {
  v16h     v;
  _Float16 h[16];
  u32x4    q4[2];
};

// 16-bit A/B fragment K index for element e of lane (half = lane/16):
// lanes 0-15: VGPR0-3 -> K 0..7, VGPR4-7 -> K 16..23 ; lanes 16-31: +8
__device__ __forceinline__ int frag_k(int e, int half) {
  return e + ((e >= 8) ? 8 : 0) + 8 * half;
}

// ---------------------------------------------------------------------------
// Weight repack: logical [K x N] f32 -> f16, WMMA-B fragment order:
// dst[(((ktile*Ntiles)+ntile)*32 + lane)*16 + e]
// ---------------------------------------------------------------------------
__global__ void k_repack_mlp(const float* __restrict__ src, _Float16* __restrict__ dst,
                             int K, int N, int total) {
  for (int tid = blockIdx.x * blockDim.x + threadIdx.x; tid < total;
       tid += gridDim.x * blockDim.x) {
    int e = tid & 15, lane = (tid >> 4) & 31, ft = tid >> 9;
    int ntiles = N >> 4;
    int ktile = ft / ntiles, ntile = ft % ntiles;
    int half = lane >> 4, n = lane & 15;
    int k = ktile * 32 + frag_k(e, half);
    dst[tid] = (_Float16)src[(size_t)k * N + ntile * 16 + n];
  }
}

// Conv weights OIHW [OC][IC][3][3] -> logical K=(tap*IC+ic), N=oc fragment order
__global__ void k_repack_conv(const float* __restrict__ w, _Float16* __restrict__ dst,
                              int IC, int total) {
  const int OC = DIMc;
  for (int tid = blockIdx.x * blockDim.x + threadIdx.x; tid < total;
       tid += gridDim.x * blockDim.x) {
    int e = tid & 15, lane = (tid >> 4) & 31, ft = tid >> 9;
    int ntiles = OC >> 4;
    int ktile = ft / ntiles, ntile = ft % ntiles;
    int half = lane >> 4, n = lane & 15;
    int k = ktile * 32 + frag_k(e, half);
    int tap = k / IC, ic = k % IC;
    int oc = ntile * 16 + n;
    dst[tid] = (_Float16)w[((size_t)oc * IC + ic) * 9 + tap];
  }
}

// Fold rel_cell rows of m0w into an effective per-batch bias.
__global__ void k_bias0(const float* __restrict__ cell, const float* __restrict__ m0w,
                        const float* __restrict__ m0b, float* __restrict__ bias0) {
  int tid = blockIdx.x * blockDim.x + threadIdx.x;
  if (tid >= B_ * HID) return;
  int b = tid / HID, n = tid % HID;
  float ry = cell[b * 2 + 0] * (float)Hh;
  float rx = cell[b * 2 + 1] * (float)Ww;
  bias0[tid] = m0b[n] + ry * m0w[(size_t)9408 * HID + n] + rx * m0w[(size_t)9409 * HID + n];
}

// ---------------------------------------------------------------------------
// Encoder conv 3->64 (tiny), direct, out NHWC f16.
// ---------------------------------------------------------------------------
__global__ void k_conv_enc(const float* __restrict__ inp, const float* __restrict__ w,
                           const float* __restrict__ bias, _Float16* __restrict__ out) {
  int tid = blockIdx.x * blockDim.x + threadIdx.x;
  if (tid >= B_ * P_ * ENC) return;
  int oc = tid % ENC;
  int pix = tid / ENC;
  int x = pix % Ww, y = (pix / Ww) % Hh, b = pix / P_;
  float acc = bias[oc];
  for (int ci = 0; ci < 3; ++ci)
    for (int ky = 0; ky < 3; ++ky) {
      int yy = y + ky - 1;
      if (yy < 0 || yy >= Hh) continue;
      for (int kx = 0; kx < 3; ++kx) {
        int xx = x + kx - 1;
        if (xx < 0 || xx >= Ww) continue;
        acc += inp[((size_t)(b * 3 + ci) * Hh + yy) * Ww + xx] *
               w[((oc * 3 + ci) * 3 + ky) * 3 + kx];
      }
    }
  out[(size_t)pix * ENC + oc] = (_Float16)acc;
}

// ---------------------------------------------------------------------------
// Implicit-GEMM 3x3 conv, OC=192, via WMMA. 16-pixel tile x 192 channels.
// Block = 128 threads (4 waves), wave handles 3 N-tiles of 16 channels.
// ---------------------------------------------------------------------------
template <int IC, bool OUT_F32>
__global__ __launch_bounds__(128) void k_conv_wmma(const _Float16* __restrict__ in,
                                                   const _Float16* __restrict__ Bw,
                                                   const float* __restrict__ bias,
                                                   float* __restrict__ out_f,
                                                   _Float16* __restrict__ out_h) {
  __shared__ _Float16 sA[16][IC];
  const int t = threadIdx.x;
  const int wave = t >> 5, lane = t & 31;
  const int half = lane >> 4, lm = lane & 15;
  const int bid = blockIdx.x;
  const int b = bid / (Hh * (Ww / 16));
  const int rem = bid % (Hh * (Ww / 16));
  const int y = rem / (Ww / 16);
  const int x0 = (rem % (Ww / 16)) * 16;
  const int NTILES = DIMc / 16;  // 12
  const int p = t >> 3, seg = t & 7;
  const int CPS = IC / 8;

  v8f acc[3];
  for (int j = 0; j < 3; ++j)
    for (int e8 = 0; e8 < 8; ++e8) acc[j][e8] = 0.f;

  for (int tap = 0; tap < 9; ++tap) {
    int dy = tap / 3 - 1, dx = tap % 3 - 1;
    int yy = y + dy;
    int xx = x0 + p + dx;
    bool inb = (yy >= 0 && yy < Hh && xx >= 0 && xx < Ww);
    const _Float16* src = in + ((size_t)(b * Hh + yy) * Ww + xx) * IC;
    for (int c = seg * CPS; c < (seg + 1) * CPS; ++c)
      sA[p][c] = inb ? src[c] : (_Float16)0.f;
    __syncthreads();
    for (int kc = 0; kc < IC / 32; ++kc) {
      FragH a;
#pragma unroll
      for (int e = 0; e < 16; ++e) a.h[e] = sA[lm][kc * 32 + frag_k(e, half)];
      int ktg = tap * (IC / 32) + kc;
#pragma unroll
      for (int j = 0; j < 3; ++j) {
        int nt = wave * 3 + j;
        const _Float16* bp = Bw + (((size_t)ktg * NTILES + nt) * 32 + lane) * 16;
        FragH bf;
        bf.q4[0] = ((const u32x4*)bp)[0];
        bf.q4[1] = ((const u32x4*)bp)[1];
        acc[j] = __builtin_amdgcn_wmma_f32_16x16x32_f16(false, a.v, false, bf.v,
                                                        (short)0, acc[j], false, false);
      }
    }
    __syncthreads();
  }

  for (int j = 0; j < 3; ++j) {
    int oc0 = (wave * 3 + j) * 16;
#pragma unroll
    for (int rr = 0; rr < 8; ++rr) {
      int m = rr + 8 * half;  // D layout: VGPR rr -> M = rr (+8 for high lanes)
      float vsum = acc[j][rr] + bias[oc0 + lm];
      size_t o = ((size_t)(b * Hh + y) * Ww + (x0 + m)) * DIMc + oc0 + lm;
      if (OUT_F32) out_f[o] = vsum;
      else         out_h[o] = (_Float16)vsum;
    }
  }
}

// ---------------------------------------------------------------------------
// Fused: bilinear q-sample, windowed nearest k/v, softmax attention,
// attention-weighted m0 GEMM (K=9408) + m1..m3 (256x256) + m4 head + base.
// One block = 16 queries, 128 threads (4 waves), each wave = 4 N-tiles.
// ---------------------------------------------------------------------------
__global__ __launch_bounds__(128) void k_attn_mlp(
    const float* __restrict__ coords, const float* __restrict__ featq,
    const float* __restrict__ featk, const float* __restrict__ featv,
    const _Float16* __restrict__ Bm0, const _Float16* __restrict__ Bm1,
    const _Float16* __restrict__ Bm2, const _Float16* __restrict__ Bm3,
    const float* __restrict__ bias0, const float* __restrict__ m1b,
    const float* __restrict__ m2b, const float* __restrict__ m3b,
    const float* __restrict__ m4w, const float* __restrict__ m4b,
    const float* __restrict__ inp, float* __restrict__ out) {
  __shared__ float    sQ[16][DIMc];
  __shared__ float    sAttn[16][RAx][HEADS];
  __shared__ _Float16 sA[16][DIMc];
  __shared__ _Float16 sAct[2][16][HID];
  __shared__ float    sCy[16], sCx[16];
  __shared__ int      sIy[16], sIx[16];

  const int t = threadIdx.x;
  const int wave = t >> 5, lane = t & 31, half = lane >> 4, lm = lane & 15;
  const int b = blockIdx.x / (Qn / 16);
  const int q0 = (blockIdx.x % (Qn / 16)) * 16;

  if (t < 16) {
    float cy = coords[((size_t)b * Qn + q0 + t) * 2 + 0];
    float cx = coords[((size_t)b * Qn + q0 + t) * 2 + 1];
    sCy[t] = cy; sCx[t] = cx;
    int iy = (int)floorf((cy + 1.f) * 32.f);  // floor(pix(cy)+0.5)
    int ix = (int)floorf((cx + 1.f) * 32.f);
    sIy[t] = min(max(iy, 0), Hh - 1);
    sIx[t] = min(max(ix, 0), Ww - 1);
  }
  __syncthreads();

  {  // bilinear q sample (zero padding), thread covers one head of one query
    int qi = t >> 3, part = t & 7;
    int c0 = part * HD;
    float py = (sCy[qi] + 1.f) * 32.f - 0.5f;
    float px = (sCx[qi] + 1.f) * 32.f - 0.5f;
    float y0f = floorf(py), x0f = floorf(px);
    float wy = py - y0f, wx = px - x0f;
    int y0 = (int)y0f, x0i = (int)x0f;
    float acc[HD];
#pragma unroll
    for (int d = 0; d < HD; ++d) acc[d] = 0.f;
    for (int dy = 0; dy < 2; ++dy)
      for (int dx = 0; dx < 2; ++dx) {
        int iy = y0 + dy, ix = x0i + dx;
        if (iy < 0 || iy >= Hh || ix < 0 || ix >= Ww) continue;
        float wgt = (dy ? wy : 1.f - wy) * (dx ? wx : 1.f - wx);
        const float* src = featq + ((size_t)(b * Hh + iy) * Ww + ix) * DIMc + c0;
#pragma unroll
        for (int d = 0; d < HD; ++d) acc[d] += src[d] * wgt;
      }
#pragma unroll
    for (int d = 0; d < HD; ++d) sQ[qi][c0 + d] = acc[d];
  }
  __syncthreads();

  {  // attention logits; OOB window samples contribute logit 0 (matches ref)
    int qi = t >> 3, rp = t & 7;
    const float scale = 0.20412414523193154f;  // 1/sqrt(24)
    for (int r = rp; r < RAx; r += 8) {
      int dy = r / RRr - RAD, dx = r % RRr - RAD;
      int yy = sIy[qi] + dy, xx = sIx[qi] + dx;
      float dot[HEADS];
#pragma unroll
      for (int h = 0; h < HEADS; ++h) dot[h] = 0.f;
      if (yy >= 0 && yy < Hh && xx >= 0 && xx < Ww) {
        const float* src = featk + ((size_t)(b * Hh + yy) * Ww + xx) * DIMc;
        for (int h = 0; h < HEADS; ++h) {
          float s = 0.f;
#pragma unroll
          for (int d = 0; d < HD; ++d) s += src[h * HD + d] * sQ[qi][h * HD + d];
          dot[h] = s;
        }
      }
      for (int h = 0; h < HEADS; ++h) sAttn[qi][r][h] = dot[h] * scale;
    }
  }
  __syncthreads();

  {  // softmax over the 49-sample window, per (query, head)
    int qi = t >> 3, h = t & 7;
    float mx = -1e30f;
    for (int r = 0; r < RAx; ++r) mx = fmaxf(mx, sAttn[qi][r][h]);
    float sum = 0.f;
    for (int r = 0; r < RAx; ++r) {
      float e = __expf(sAttn[qi][r][h] - mx);
      sAttn[qi][r][h] = e;
      sum += e;
    }
    float inv = 1.f / sum;
    for (int r = 0; r < RAx; ++r) sAttn[qi][r][h] *= inv;
  }
  __syncthreads();

  // ----- m0 GEMM fused with attention weighting: K = 49*192 = 9408 -----
  v8f acc0[4];
  for (int j = 0; j < 4; ++j)
    for (int e8 = 0; e8 < 8; ++e8) acc0[j][e8] = 0.f;
  {
    int p = t >> 3, seg = t & 7;  // query, head
    for (int r = 0; r < RAx; ++r) {
      int dy = r / RRr - RAD, dx = r % RRr - RAD;
      int yy = sIy[p] + dy, xx = sIx[p] + dx;
      bool inb = (yy >= 0 && yy < Hh && xx >= 0 && xx < Ww);
      float aw = sAttn[p][r][seg];
      const float* src = featv + ((size_t)(b * Hh + yy) * Ww + xx) * DIMc + seg * HD;
#pragma unroll
      for (int d = 0; d < HD; ++d) {
        float val = inb ? src[d] * aw : 0.f;
        sA[p][seg * HD + d] = (_Float16)val;
      }
      __syncthreads();
      for (int kc = 0; kc < DIMc / 32; ++kc) {
        FragH a;
#pragma unroll
        for (int e = 0; e < 16; ++e) a.h[e] = sA[lm][kc * 32 + frag_k(e, half)];
        int ktg = r * (DIMc / 32) + kc;
#pragma unroll
        for (int j = 0; j < 4; ++j) {
          int nt = wave * 4 + j;
          const _Float16* bp = Bm0 + (((size_t)ktg * 16 + nt) * 32 + lane) * 16;
          FragH bf;
          bf.q4[0] = ((const u32x4*)bp)[0];
          bf.q4[1] = ((const u32x4*)bp)[1];
          acc0[j] = __builtin_amdgcn_wmma_f32_16x16x32_f16(false, a.v, false, bf.v,
                                                           (short)0, acc0[j], false, false);
        }
      }
      __syncthreads();
    }
  }
  for (int j = 0; j < 4; ++j) {  // bias (incl rel_cell) + ReLU -> act0
    int nc = (wave * 4 + j) * 16 + lm;
    float bb = bias0[b * HID + nc];
#pragma unroll
    for (int rr = 0; rr < 8; ++rr) {
      int m = rr + 8 * half;
      float vsum = acc0[j][rr] + bb;
      sAct[0][m][nc] = (_Float16)(vsum > 0.f ? vsum : 0.f);
    }
  }
  __syncthreads();

  // ----- hidden layers m1..m3: 16x256 @ 256x256, ping-pong in LDS -----
  for (int L = 0; L < 3; ++L) {
    const _Float16* Wl = (L == 0) ? Bm1 : (L == 1) ? Bm2 : Bm3;
    const float* bl = (L == 0) ? m1b : (L == 1) ? m2b : m3b;
    const _Float16(*src)[HID] = sAct[L & 1];
    _Float16(*dst)[HID] = sAct[(L + 1) & 1];
    v8f acc[4];
    for (int j = 0; j < 4; ++j)
      for (int e8 = 0; e8 < 8; ++e8) acc[j][e8] = 0.f;
    for (int kt = 0; kt < HID / 32; ++kt) {
      FragH a;
#pragma unroll
      for (int e = 0; e < 16; ++e) a.h[e] = src[lm][kt * 32 + frag_k(e, half)];
#pragma unroll
      for (int j = 0; j < 4; ++j) {
        int nt = wave * 4 + j;
        const _Float16* bp = Wl + (((size_t)kt * 16 + nt) * 32 + lane) * 16;
        FragH bf;
        bf.q4[0] = ((const u32x4*)bp)[0];
        bf.q4[1] = ((const u32x4*)bp)[1];
        acc[j] = __builtin_amdgcn_wmma_f32_16x16x32_f16(false, a.v, false, bf.v,
                                                        (short)0, acc[j], false, false);
      }
    }
    __syncthreads();
    for (int j = 0; j < 4; ++j) {
      int nc = (wave * 4 + j) * 16 + lm;
      float bb = bl[nc];
#pragma unroll
      for (int rr = 0; rr < 8; ++rr) {
        int m = rr + 8 * half;
        float vsum = acc[j][rr] + bb;
        dst[m][nc] = (_Float16)(vsum > 0.f ? vsum : 0.f);
      }
    }
    __syncthreads();
  }

  // ----- m4 (256->3) + border-bilinear base -----
  if (t < 48) {
    int qi = t / 3, o = t % 3;
    float dot = m4b[o];
    for (int k = 0; k < HID; ++k) dot += (float)sAct[1][qi][k] * m4w[k * 3 + o];
    float py = (sCy[qi] + 1.f) * 32.f - 0.5f;
    float px = (sCx[qi] + 1.f) * 32.f - 0.5f;
    float y0f = floorf(py), x0f = floorf(px);
    float wy = py - y0f, wx = px - x0f;
    int y0 = (int)y0f, x0i = (int)x0f;
    float base = 0.f;
    for (int dy = 0; dy < 2; ++dy)
      for (int dx = 0; dx < 2; ++dx) {
        int iy = min(max(y0 + dy, 0), Hh - 1);
        int ix = min(max(x0i + dx, 0), Ww - 1);
        float wgt = (dy ? wy : 1.f - wy) * (dx ? wx : 1.f - wx);
        base += inp[((size_t)(b * 3 + o) * Hh + iy) * Ww + ix] * wgt;
      }
    out[((size_t)b * Qn + q0 + qi) * 3 + o] = dot + base;
  }
}

// ---------------------------------------------------------------------------
extern "C" void kernel_launch(void* const* d_in, const int* in_sizes, int n_in,
                              void* d_out, int out_size, void* d_ws, size_t ws_size,
                              hipStream_t stream) {
  (void)in_sizes; (void)n_in; (void)out_size; (void)ws_size;
  const float* inp    = (const float*)d_in[0];
  const float* coords = (const float*)d_in[1];
  const float* cell   = (const float*)d_in[2];
  const float* enc_w  = (const float*)d_in[3];
  const float* enc_b  = (const float*)d_in[4];
  const float* ch_w   = (const float*)d_in[5];
  const float* ch_b   = (const float*)d_in[6];
  const float* q_w    = (const float*)d_in[7];
  const float* q_b    = (const float*)d_in[8];
  const float* k_w    = (const float*)d_in[9];
  const float* k_b    = (const float*)d_in[10];
  const float* v_w    = (const float*)d_in[11];
  const float* v_b    = (const float*)d_in[12];
  const float* m0w    = (const float*)d_in[13];
  const float* m0b    = (const float*)d_in[14];
  const float* m1w    = (const float*)d_in[15];
  const float* m1b    = (const float*)d_in[16];
  const float* m2w    = (const float*)d_in[17];
  const float* m2b    = (const float*)d_in[18];
  const float* m3w    = (const float*)d_in[19];
  const float* m3b    = (const float*)d_in[20];
  const float* m4w    = (const float*)d_in[21];
  const float* m4b    = (const float*)d_in[22];
  float* out = (float*)d_out;

  // Workspace carve-up (~30.5 MB total)
  char* ws = (char*)d_ws;
  size_t off = 0;
  auto carve = [&](size_t bytes) -> char* {
    char* p = ws + off;
    off += (bytes + 255) & ~(size_t)255;
    return p;
  };
  _Float16* feat0 = (_Float16*)carve((size_t)B_ * P_ * ENC * 2);
  _Float16* feat  = (_Float16*)carve((size_t)B_ * P_ * DIMc * 2);
  float* featq    = (float*)carve((size_t)B_ * P_ * DIMc * 4);
  float* featk    = (float*)carve((size_t)B_ * P_ * DIMc * 4);
  float* featv    = (float*)carve((size_t)B_ * P_ * DIMc * 4);
  _Float16* Bch   = (_Float16*)carve((size_t)576 * 192 * 2);
  _Float16* Bq    = (_Float16*)carve((size_t)1728 * 192 * 2);
  _Float16* Bk    = (_Float16*)carve((size_t)1728 * 192 * 2);
  _Float16* Bv    = (_Float16*)carve((size_t)1728 * 192 * 2);
  _Float16* Bm0   = (_Float16*)carve((size_t)9408 * 256 * 2);
  _Float16* Bm1   = (_Float16*)carve((size_t)256 * 256 * 2);
  _Float16* Bm2   = (_Float16*)carve((size_t)256 * 256 * 2);
  _Float16* Bm3   = (_Float16*)carve((size_t)256 * 256 * 2);
  float* bias0    = (float*)carve((size_t)B_ * HID * 4);

  // ---- weight repacks (independent) ----
  {
    int tot = 576 * 192;
    k_repack_conv<<<(tot + 255) / 256, 256, 0, stream>>>(ch_w, Bch, 64, tot);
  }
  {
    int tot = 1728 * 192;
    int g = (tot + 255) / 256;
    k_repack_conv<<<g, 256, 0, stream>>>(q_w, Bq, 192, tot);
    k_repack_conv<<<g, 256, 0, stream>>>(k_w, Bk, 192, tot);
    k_repack_conv<<<g, 256, 0, stream>>>(v_w, Bv, 192, tot);
  }
  {
    int tot = 9408 * 256;
    k_repack_mlp<<<(tot + 255) / 256, 256, 0, stream>>>(m0w, Bm0, 9408, 256, tot);
  }
  {
    int tot = 256 * 256;
    int g = (tot + 255) / 256;
    k_repack_mlp<<<g, 256, 0, stream>>>(m1w, Bm1, 256, 256, tot);
    k_repack_mlp<<<g, 256, 0, stream>>>(m2w, Bm2, 256, 256, tot);
    k_repack_mlp<<<g, 256, 0, stream>>>(m3w, Bm3, 256, 256, tot);
  }
  k_bias0<<<2, 256, 0, stream>>>(cell, m0w, m0b, bias0);

  // ---- feature pipeline ----
  k_conv_enc<<<(B_ * P_ * ENC + 255) / 256, 256, 0, stream>>>(inp, enc_w, enc_b, feat0);

  const int convGrid = B_ * Hh * (Ww / 16);  // 512
  k_conv_wmma<64, false><<<convGrid, 128, 0, stream>>>(feat0, Bch, ch_b, nullptr, feat);
  k_conv_wmma<192, true><<<convGrid, 128, 0, stream>>>(feat, Bq, q_b, featq, nullptr);
  k_conv_wmma<192, true><<<convGrid, 128, 0, stream>>>(feat, Bk, k_b, featk, nullptr);
  k_conv_wmma<192, true><<<convGrid, 128, 0, stream>>>(feat, Bv, v_b, featv, nullptr);

  // ---- fused attention + MLP ----
  k_attn_mlp<<<B_ * (Qn / 16), 128, 0, stream>>>(coords, featq, featk, featv, Bm0, Bm1,
                                                 Bm2, Bm3, bias0, m1b, m2b, m3b, m4w,
                                                 m4b, inp, out);
}
